// SeqAttnMatch_15247133901008
// MI455X (gfx1250) — matrix operand order repaired and checked by reading, not compiled, not run on hardware
//
#include <hip/hip_runtime.h>
#include <hip/hip_bf16.h>
#include <stdint.h>

// ---------------------------------------------------------------------------
// SeqAttnMatch on MI455X (gfx1250), bf16 WMMA pipeline, L2-resident intermediates
// B=32, L1=L2=512, H=768
// ---------------------------------------------------------------------------

#define BB   32
#define LL   512
#define HH   768

typedef __attribute__((ext_vector_type(8)))  float   v8f;
typedef __attribute__((ext_vector_type(16))) __bf16  v16bf;

union Frag16 {            // 16 bf16 = 32 bytes = one WMMA A/B fragment per lane
    v16bf v;
    uint4 q[2];
};

// float -> bf16 bits, round-to-nearest-even (manual: avoids scalar __bf16 math)
__device__ __forceinline__ unsigned short f2bf(float f) {
    unsigned u = __builtin_bit_cast(unsigned, f);
    u += 0x7FFFu + ((u >> 16) & 1u);
    return (unsigned short)(u >> 16);
}

// A fragment (16x32 bf16): lane holds row m = mBase + (lane&15),
// K chunks at k0 + 8*(lane>>4) and +16.   [CDNA5 ISA §7.12.2]
__device__ __forceinline__ v16bf load_a_frag(const unsigned short* A, int lda,
                                             int mBase, int k0, int lane) {
    int row = mBase + (lane & 15);
    int kb  = k0 + ((lane >> 4) << 3);
    const unsigned short* p = A + (size_t)row * lda + kb;
    __builtin_prefetch(p + 32, 0, 1);   // next k-tile -> global_prefetch_b8
    Frag16 t;
    t.q[0] = *(const uint4*)(p);
    t.q[1] = *(const uint4*)(p + 16);
    return t.v;
}

// B fragment (32x16 bf16), B row-major [K][N]: lane holds column n = n0 + (lane&15),
// K chunks at k0 + 8*(lane>>4) and +16 (same order as A).
__device__ __forceinline__ v16bf load_b_frag(const unsigned short* Bm, int ldb,
                                             int k0, int n0, int lane) {
    const unsigned short* p = Bm + (size_t)(k0 + ((lane >> 4) << 3)) * ldb + n0 + (lane & 15);
    __builtin_prefetch(p + 32 * ldb, 0, 1);
    Frag16 t;
    unsigned short e[16];
#pragma unroll
    for (int s = 0; s < 16; ++s)
        e[s] = p[(size_t)(s < 8 ? s : s + 8) * ldb];
    __builtin_memcpy(&t, e, sizeof(e));
    return t.v;
}

// One wave computes a 32(M) x 64(N) tile: 2 A-frags x 4 B-frags, K-loop step 32.
__device__ __forceinline__ void wave_gemm_32x64(const unsigned short* A, int lda,
                                                const unsigned short* Bm, int ldb,
                                                int K, int mBase, int nBase,
                                                int lane, v8f acc[2][4]) {
    for (int k0 = 0; k0 < K; k0 += 32) {
        v16bf a[2], b[4];
#pragma unroll
        for (int i = 0; i < 2; ++i)
            a[i] = load_a_frag(A, lda, mBase + i * 16, k0, lane);
#pragma unroll
        for (int j = 0; j < 4; ++j)
            b[j] = load_b_frag(Bm, ldb, k0, nBase + j * 16, lane);
#pragma unroll
        for (int i = 0; i < 2; ++i)
#pragma unroll
            for (int j = 0; j < 4; ++j)
                acc[i][j] = __builtin_amdgcn_wmma_f32_16x16x32_bf16(
                    false, a[i], false, b[j], (short)0, acc[i][j], false, false);
    }
}

// ---------------------------------------------------------------------------
// Kernel 1a: f32 -> bf16 bulk convert (x and y), 4 elements / thread
// ---------------------------------------------------------------------------
__global__ __launch_bounds__(256) void cvt_bf16_kernel(const float* __restrict__ src,
                                                       unsigned short* __restrict__ dst,
                                                       int n) {
    int idx = (blockIdx.x * blockDim.x + threadIdx.x) * 4;
    if (idx >= n) return;
    float4 f = *(const float4*)(src + idx);
    uint2 o;
    o.x = (unsigned)f2bf(f.x) | ((unsigned)f2bf(f.y) << 16);
    o.y = (unsigned)f2bf(f.z) | ((unsigned)f2bf(f.w) << 16);
    *(uint2*)(dst + idx) = o;
}

// Kernel 1b: Wt[k][n] = bf16(W[n][k])  (W is nn.Linear [out,in])
__global__ __launch_bounds__(256) void wt_kernel(const float* __restrict__ W,
                                                 unsigned short* __restrict__ Wt) {
    int idx = blockIdx.x * blockDim.x + threadIdx.x;
    if (idx >= HH * HH) return;
    int k = idx / HH, n = idx % HH;
    Wt[idx] = f2bf(W[(size_t)n * HH + k]);
}

// ---------------------------------------------------------------------------
// Kernel 2: shared projection  P = relu(Z @ Wt + b)
//   part 0: Z = xb (B*L1 rows) -> xp   [B*L1][H]   row-major
//   part 1: Z = yb (B*L2 rows) -> ypT  [B][H][L2]  (stored transposed!)
// grid (128, 6, 2), block 256 (8 waves, 4Mx2N), wg tile 128x128
// ---------------------------------------------------------------------------
__global__ __launch_bounds__(256)
void proj_kernel(const unsigned short* __restrict__ xb,
                 const unsigned short* __restrict__ yb,
                 const unsigned short* __restrict__ Wt,
                 const float* __restrict__ bias,
                 unsigned short* __restrict__ xp,
                 unsigned short* __restrict__ ypT) {
    int wave = threadIdx.x >> 5, lane = threadIdx.x & 31;
    int wm = wave & 3, wn = wave >> 2;
    int part = blockIdx.z;
    const unsigned short* A = part ? yb : xb;
    int mOff = blockIdx.x * 128 + wm * 32;
    int nOff = blockIdx.y * 128 + wn * 64;

    v8f acc[2][4] = {};
    wave_gemm_32x64(A, HH, Wt, HH, HH, mOff, nOff, lane, acc);

    int ln = lane & 15, lh = lane >> 4;
#pragma unroll
    for (int i = 0; i < 2; ++i) {
        int rbase = mOff + i * 16 + lh * 8;       // 8 consecutive M rows
#pragma unroll
        for (int j = 0; j < 4; ++j) {
            int col = nOff + j * 16 + ln;         // output feature
            float bv = bias[col];
            if (part == 0) {
#pragma unroll
                for (int r = 0; r < 8; ++r) {
                    float v = acc[i][j][r] + bv;
                    v = v > 0.f ? v : 0.f;
                    xp[(size_t)(rbase + r) * HH + col] = f2bf(v);
                }
            } else {
                // transposed store: ypT[b][col][n2..n2+7] contiguous (16B)
                int bb = rbase >> 9, n2 = rbase & (LL - 1);
                unsigned pk[4];
#pragma unroll
                for (int t = 0; t < 4; ++t) {
                    float v0 = acc[i][j][2 * t] + bv;     v0 = v0 > 0.f ? v0 : 0.f;
                    float v1 = acc[i][j][2 * t + 1] + bv; v1 = v1 > 0.f ? v1 : 0.f;
                    pk[t] = (unsigned)f2bf(v0) | ((unsigned)f2bf(v1) << 16);
                }
                *(uint4*)(ypT + ((size_t)bb * HH + col) * LL + n2) =
                    make_uint4(pk[0], pk[1], pk[2], pk[3]);
            }
        }
    }
}

// ---------------------------------------------------------------------------
// Kernel 3: scores[b] = xp[b] (512x768) @ ypT[b] (768x512)   -> f32
// grid (4, 4, 32)
// ---------------------------------------------------------------------------
__global__ __launch_bounds__(256)
void scores_kernel(const unsigned short* __restrict__ xp,
                   const unsigned short* __restrict__ ypT,
                   float* __restrict__ scores) {
    int wave = threadIdx.x >> 5, lane = threadIdx.x & 31;
    int wm = wave & 3, wn = wave >> 2;
    int b = blockIdx.z;
    const unsigned short* A  = xp  + (size_t)b * LL * HH;
    const unsigned short* Bm = ypT + (size_t)b * HH * LL;
    float* C = scores + (size_t)b * LL * LL;
    int mOff = blockIdx.x * 128 + wm * 32;
    int nOff = blockIdx.y * 128 + wn * 64;

    v8f acc[2][4] = {};
    wave_gemm_32x64(A, HH, Bm, LL, HH, mOff, nOff, lane, acc);

    int ln = lane & 15, lh = lane >> 4;
#pragma unroll
    for (int i = 0; i < 2; ++i) {
        int rbase = mOff + i * 16 + lh * 8;
#pragma unroll
        for (int j = 0; j < 4; ++j) {
            int col = nOff + j * 16 + ln;
#pragma unroll
            for (int r = 0; r < 8; ++r)
                C[(size_t)(rbase + r) * LL + col] = acc[i][j][r];
        }
    }
}

// ---------------------------------------------------------------------------
// Kernel 4: masked softmax over L2; one wave per row (wave32, 16 cols/lane)
// writes alpha as bf16. grid 2048, block 256 (8 rows per block)
// ---------------------------------------------------------------------------
__global__ __launch_bounds__(256)
void softmax_kernel(const float* __restrict__ scores,
                    const unsigned char* __restrict__ ymask,
                    unsigned short* __restrict__ alpha) {
    int wave = threadIdx.x >> 5, lane = threadIdx.x & 31;
    int row = blockIdx.x * 8 + wave;              // [0, B*L1)
    int b = row >> 9;
    const float* s = scores + (size_t)row * LL;
    const unsigned char* mk = ymask + (size_t)b * LL;

    float v[16];
    float mx = -INFINITY;
#pragma unroll
    for (int t = 0; t < 16; ++t) {
        int c = lane + 32 * t;
        float x = mk[c] ? -INFINITY : s[c];
        v[t] = x;
        mx = fmaxf(mx, x);
    }
#pragma unroll
    for (int off = 16; off > 0; off >>= 1)
        mx = fmaxf(mx, __shfl_xor(mx, off, 32));

    float sum = 0.f;
#pragma unroll
    for (int t = 0; t < 16; ++t) {
        float e = __expf(v[t] - mx);
        v[t] = e;
        sum += e;
    }
#pragma unroll
    for (int off = 16; off > 0; off >>= 1)
        sum += __shfl_xor(sum, off, 32);

    float inv = 1.0f / sum;
    unsigned short* a = alpha + (size_t)row * LL;
#pragma unroll
    for (int t = 0; t < 16; ++t)
        a[lane + 32 * t] = f2bf(v[t] * inv);
}

// ---------------------------------------------------------------------------
// Kernel 5: matched[b] = alpha[b] (512x512) @ yb[b] (512x768)  -> f32 out
// grid (4, 6, 32)
// ---------------------------------------------------------------------------
__global__ __launch_bounds__(256)
void matched_kernel(const unsigned short* __restrict__ alpha,
                    const unsigned short* __restrict__ yb,
                    float* __restrict__ out) {
    int wave = threadIdx.x >> 5, lane = threadIdx.x & 31;
    int wm = wave & 3, wn = wave >> 2;
    int b = blockIdx.z;
    const unsigned short* A  = alpha + (size_t)b * LL * LL;
    const unsigned short* Bm = yb    + (size_t)b * LL * HH;
    float* C = out + (size_t)b * LL * HH;
    int mOff = blockIdx.x * 128 + wm * 32;
    int nOff = blockIdx.y * 128 + wn * 64;

    v8f acc[2][4] = {};
    wave_gemm_32x64(A, LL, Bm, HH, LL, mOff, nOff, lane, acc);

    int ln = lane & 15, lh = lane >> 4;
#pragma unroll
    for (int i = 0; i < 2; ++i) {
        int rbase = mOff + i * 16 + lh * 8;
#pragma unroll
        for (int j = 0; j < 4; ++j) {
            int col = nOff + j * 16 + ln;
#pragma unroll
            for (int r = 0; r < 8; ++r)
                C[(size_t)(rbase + r) * HH + col] = acc[i][j][r];
        }
    }
}

// ---------------------------------------------------------------------------
// host launcher
// ---------------------------------------------------------------------------
extern "C" void kernel_launch(void* const* d_in, const int* in_sizes, int n_in,
                              void* d_out, int out_size, void* d_ws, size_t ws_size,
                              hipStream_t stream) {
    const float*         x     = (const float*)d_in[0];       // [B,L1,H] f32
    const float*         y     = (const float*)d_in[1];       // [B,L2,H] f32
    const unsigned char* ymask = (const unsigned char*)d_in[2]; // [B,L2] bool
    const float*         W     = (const float*)d_in[3];       // [H,H] f32
    const float*         bias  = (const float*)d_in[4];       // [H] f32
    float*               out   = (float*)d_out;               // [B,L1,H] f32

    const size_t nXY = (size_t)BB * LL * HH;   // 12,582,912
    char* ws = (char*)d_ws;
    size_t off = 0;
    unsigned short* xb  = (unsigned short*)(ws + off); off += nXY * 2;           // 25.2 MB
    unsigned short* yb  = (unsigned short*)(ws + off); off += nXY * 2;           // 25.2 MB
    unsigned short* Wt  = (unsigned short*)(ws + off); off += (size_t)HH * HH * 2; // 1.2 MB
    unsigned short* xp  = (unsigned short*)(ws + off); off += nXY * 2;           // 25.2 MB
    unsigned short* ypT = (unsigned short*)(ws + off); off += nXY * 2;           // 25.2 MB
    float*          sc  = (float*)(ws + off);          off += (size_t)BB * LL * LL * 4; // 33.6 MB
    unsigned short* al  = (unsigned short*)(ws + off); off += (size_t)BB * LL * LL * 2; // 16.8 MB

    // 1) precision convert + weight transpose (bandwidth pass, stays in L2)
    cvt_bf16_kernel<<<(int)(nXY / 4 + 255) / 256, 256, 0, stream>>>(x, xb, (int)nXY);
    cvt_bf16_kernel<<<(int)(nXY / 4 + 255) / 256, 256, 0, stream>>>(y, yb, (int)nXY);
    wt_kernel<<<(HH * HH + 255) / 256, 256, 0, stream>>>(W, Wt);

    // 2) shared projection (x -> xp row-major, y -> ypT transposed)
    proj_kernel<<<dim3(128, 6, 2), 256, 0, stream>>>(xb, yb, Wt, bias, xp, ypT);

    // 3) attention scores (bf16 WMMA, f32 out)
    scores_kernel<<<dim3(4, 4, 32), 256, 0, stream>>>(xp, ypT, sc);

    // 4) masked softmax -> bf16 alpha
    softmax_kernel<<<2048, 256, 0, stream>>>(sc, ymask, al);

    // 5) weighted sum over original y -> f32 output
    matched_kernel<<<dim3(4, 6, 32), 256, 0, stream>>>(al, yb, out);
}
